// GCNModifiedAE_72756745994507
// MI455X (gfx1250) — matrix operand order repaired and checked by reading, not compiled, not run on hardware
//
#include <hip/hip_runtime.h>

// ---------------------------------------------------------------------------
// GCN autoencoder layer for MI455X (gfx1250, wave32, WMMA):
//   Wpack   : one-shot f16 repack of W into B-fragment-major layouts
//   support = x @ W            (v_wmma_f32_16x16x32_f16, b128 clauses)
//   hidden  = scatter-add over edges (f32 atomics, L2-resident)
//   adj     = hidden @ W^T     (v_wmma_f32_16x16x32_f16)
// d_out = [adj (50000x512) | hidden (50000x256)] concatenated.
// B fragments for a k-step are preloaded as a batch so the compiler can clause
// the loads and overlap their latency with the WMMA chain.
// ---------------------------------------------------------------------------

#define N_NODES 50000
#define F_IN    512
#define F_HID   256

typedef __attribute__((ext_vector_type(16))) _Float16 v16h;
typedef __attribute__((ext_vector_type(8)))  float    v8f;

// ---------------------------------------------------------------------------
// Zero the hidden accumulator region (float4-vectorized, grid-stride).
// ---------------------------------------------------------------------------
__global__ void __launch_bounds__(256) zero_f4_kernel(float4* __restrict__ p, int n4) {
    int i      = blockIdx.x * blockDim.x + threadIdx.x;
    int stride = gridDim.x * blockDim.x;
    for (; i < n4; i += stride) p[i] = make_float4(0.f, 0.f, 0.f, 0.f);
}

// ---------------------------------------------------------------------------
// Pack W (f32 [512,256]) into two f16 tables so B fragments are contiguous:
//   ws16[n*512 + k] = f16(W[k][n])   -> encoder B (k-major per n)
//   wd16[n*256 + k] = f16(W[n][k])   -> decoder B (same layout as weight)
// ---------------------------------------------------------------------------
__global__ void __launch_bounds__(256)
pack_w_kernel(const float* __restrict__ w, _Float16* __restrict__ ws16,
              _Float16* __restrict__ wd16) {
    const int idx = blockIdx.x * 256 + threadIdx.x;  // 0 .. 131071
    const int r   = idx >> 8;   // row in weight  (k for encoder, n for decoder)
    const int c   = idx & 255;  // col in weight  (n for encoder, k for decoder)
    const _Float16 h = (_Float16)w[idx];
    wd16[idx]                  = h;
    ws16[(size_t)c * F_IN + r] = h;
}

// ---------------------------------------------------------------------------
// support[M=50000, N=256] = x[M, K=512] @ W[K, N]
// Block = 128 threads (4 waves). Block b owns M-rows [16b, 16b+16).
// Wave w owns N-cols [64w, 64w+64) as four 16x16 WMMA tiles.
// Per k-step: A = 4x b128 + cvt_pk (reused by all 4 tiles); all 4 B fragments
// preloaded (8x b128, one clause) before the 4-WMMA chain.
// Fragment k order: contiguous, k = 16*hi + s in both A and B (wave32).
// ---------------------------------------------------------------------------
__global__ void __launch_bounds__(128)
gemm_support_kernel(const float* __restrict__ x, const _Float16* __restrict__ ws16,
                    float* __restrict__ support) {
    const int lane  = threadIdx.x & 31;
    const int wave  = threadIdx.x >> 5;   // 0..3
    const int l16   = lane & 15;
    const int hi    = lane >> 4;          // 0 or 1
    const int mbase = blockIdx.x * 16;    // 50000 = 3125*16, exact
    const int nbase = wave * 64;

    v8f acc[4] = {v8f{}, v8f{}, v8f{}, v8f{}};

    const float* xrow = x + (size_t)(mbase + l16) * F_IN;
    const _Float16* bcol = ws16 + (size_t)(nbase + l16) * F_IN + hi * 16;
    const int ko = hi * 16;  // A: lanes 0-15 K{0..15}; lanes 16-31 K{16..31}

    for (int k0 = 0; k0 < F_IN; k0 += 32) {
        // A fragment: two 32B vector loads + f32->f16 convert (pairs to cvt_pk)
        const v8f alo = *(const v8f*)(xrow + k0 + ko);       // 32B aligned
        const v8f ahi = *(const v8f*)(xrow + k0 + ko + 8);
        v16h a;
        #pragma unroll
        for (int i = 0; i < 8; ++i) { a[i] = (_Float16)alo[i]; a[8 + i] = (_Float16)ahi[i]; }

        // Preload all 4 B fragments (independent -> one s_clause of b128s)
        v16h b[4];
        #pragma unroll
        for (int t = 0; t < 4; ++t)
            b[t] = *(const v16h*)(bcol + (size_t)t * 16 * F_IN + k0);  // 32B aligned

        #pragma unroll
        for (int t = 0; t < 4; ++t)
            acc[t] = __builtin_amdgcn_wmma_f32_16x16x32_f16(
                false, a, false, b[t], (short)0, acc[t], false, false);
    }

    // C/D layout: VGPR r holds (row = mbase + 8*hi + r, col = nbase + 16t + l16)
    #pragma unroll
    for (int t = 0; t < 4; ++t) {
        const int col = nbase + t * 16 + l16;
        #pragma unroll
        for (int r = 0; r < 8; ++r)
            support[(size_t)(mbase + hi * 8 + r) * F_HID + col] = acc[t][r];
    }
}

// ---------------------------------------------------------------------------
// Edge scatter: hidden[dst] += support[src] * ew, one wave per edge.
// 256 floats/row = 32 lanes x 8 floats (two float4 loads). support & hidden
// are both L2-resident (51 MB each < 192 MB L2), so gathers and f32 atomics
// stay on-chip.
// ---------------------------------------------------------------------------
__global__ void __launch_bounds__(256)
spmm_scatter_kernel(const int* __restrict__ dst_idx, const int* __restrict__ src_idx,
                    const float* __restrict__ ew, const float* __restrict__ support,
                    float* __restrict__ hidden, int n_edges) {
    const int lane    = threadIdx.x & 31;
    int       wid     = blockIdx.x * (blockDim.x >> 5) + (threadIdx.x >> 5);
    const int n_waves = gridDim.x * (blockDim.x >> 5);

    for (int e = wid; e < n_edges; e += n_waves) {
        const int   d  = dst_idx[e];
        const int   s  = src_idx[e];
        const float wt = ew[e];

        const float4* sp = (const float4*)(support + (size_t)s * F_HID);
        const float4  v0 = sp[lane];        // floats [4*lane, 4*lane+4)
        const float4  v1 = sp[lane + 32];   // floats [128+4*lane, ...)

        float* hp = hidden + (size_t)d * F_HID;
        atomicAdd(hp + 4 * lane + 0,       v0.x * wt);
        atomicAdd(hp + 4 * lane + 1,       v0.y * wt);
        atomicAdd(hp + 4 * lane + 2,       v0.z * wt);
        atomicAdd(hp + 4 * lane + 3,       v0.w * wt);
        atomicAdd(hp + 128 + 4 * lane + 0, v1.x * wt);
        atomicAdd(hp + 128 + 4 * lane + 1, v1.y * wt);
        atomicAdd(hp + 128 + 4 * lane + 2, v1.z * wt);
        atomicAdd(hp + 128 + 4 * lane + 3, v1.w * wt);
    }
}

// ---------------------------------------------------------------------------
// adj[M=50000, N=512] = hidden[M, K=256] @ W^T, B[k][n] = W[n][k] = wd16[n*256+k]
// Block = 256 threads (8 waves x 64 cols = 512 = full N).
// ---------------------------------------------------------------------------
__global__ void __launch_bounds__(256)
gemm_decode_kernel(const float* __restrict__ hidden, const _Float16* __restrict__ wd16,
                   float* __restrict__ adj) {
    const int lane  = threadIdx.x & 31;
    const int wave  = threadIdx.x >> 5;   // 0..7
    const int l16   = lane & 15;
    const int hi    = lane >> 4;
    const int mbase = blockIdx.x * 16;
    const int nbase = wave * 64;

    v8f acc[4] = {v8f{}, v8f{}, v8f{}, v8f{}};

    const float* hrow = hidden + (size_t)(mbase + l16) * F_HID;
    const _Float16* bcol = wd16 + (size_t)(nbase + l16) * F_HID + hi * 16;
    const int ko = hi * 16;

    for (int k0 = 0; k0 < F_HID; k0 += 32) {
        const v8f alo = *(const v8f*)(hrow + k0 + ko);
        const v8f ahi = *(const v8f*)(hrow + k0 + ko + 8);
        v16h a;
        #pragma unroll
        for (int i = 0; i < 8; ++i) { a[i] = (_Float16)alo[i]; a[8 + i] = (_Float16)ahi[i]; }

        v16h b[4];
        #pragma unroll
        for (int t = 0; t < 4; ++t)
            b[t] = *(const v16h*)(bcol + (size_t)t * 16 * F_HID + k0);  // 32B aligned

        #pragma unroll
        for (int t = 0; t < 4; ++t)
            acc[t] = __builtin_amdgcn_wmma_f32_16x16x32_f16(
                false, a, false, b[t], (short)0, acc[t], false, false);
    }

    #pragma unroll
    for (int t = 0; t < 4; ++t) {
        const int col = nbase + t * 16 + l16;
        #pragma unroll
        for (int r = 0; r < 8; ++r)
            adj[(size_t)(mbase + hi * 8 + r) * F_IN + col] = acc[t][r];
    }
}

// ---------------------------------------------------------------------------
extern "C" void kernel_launch(void* const* d_in, const int* in_sizes, int n_in,
                              void* d_out, int out_size, void* d_ws, size_t ws_size,
                              hipStream_t stream) {
    const float* x    = (const float*)d_in[0];           // [50000, 512]
    const int*   eidx = (const int*)d_in[1];             // [2, E] (dst row, src row)
    const float* ew   = (const float*)d_in[2];           // [E]
    const float* w    = (const float*)d_in[3];           // [512, 256]
    const int    E    = in_sizes[2];                     // 1,600,000

    float* adj    = (float*)d_out;                              // [50000, 512]
    float* hidden = (float*)d_out + (size_t)N_NODES * F_IN;     // [50000, 256]

    // Workspace: [ support f32 (51.2 MB) | ws16 f16 (256 KB) | wd16 f16 (256 KB) ]
    float*    support = (float*)d_ws;
    _Float16* ws16    = (_Float16*)((char*)d_ws + (size_t)N_NODES * F_HID * sizeof(float));
    _Float16* wd16    = ws16 + (size_t)F_HID * F_IN;

    // 0) pack weights to f16 fragment-major tables
    pack_w_kernel<<<(F_IN * F_HID) / 256, 256, 0, stream>>>(w, ws16, wd16);

    // 1) hidden = 0 (atomic accumulation target; re-zeroed every call)
    zero_f4_kernel<<<2048, 256, 0, stream>>>((float4*)hidden, N_NODES * F_HID / 4);

    // 2) support = x @ W
    gemm_support_kernel<<<N_NODES / 16, 128, 0, stream>>>(x, ws16, support);

    // 3) hidden[dst] += support[src] * ew  (one wave per edge, grid-stride)
    spmm_scatter_kernel<<<2048, 256, 0, stream>>>(eidx, eidx + E, ew, support, hidden, E);

    // 4) adj = hidden @ W^T
    gemm_decode_kernel<<<N_NODES / 16, 256, 0, stream>>>(hidden, wd16, adj);
}
